// ClusterAttention_35837207118602
// MI455X (gfx1250) — matrix operand order repaired and checked by reading, not compiled, not run on hardware
//
#include <hip/hip_runtime.h>
#include <math.h>

#define RTOT 131072   // B*N rows
#define NSEQ 32768
#define BB 4
#define HH 8

typedef __attribute__((ext_vector_type(16))) __bf16 v16bf;
typedef __attribute__((ext_vector_type(8)))  float  v8f;

union V16U { v16bf v; __bf16 e[16]; uint4 q[2]; };

__device__ __forceinline__ unsigned short f2bf_raw(float f) {
  union { float f; unsigned u; } c; c.f = f;
  unsigned r = c.u + 0x7FFFu + ((c.u >> 16) & 1u);   // round-to-nearest-even
  return (unsigned short)(r >> 16);
}
__device__ __forceinline__ __bf16 f2bf(float f) {
  union { unsigned short s; __bf16 b; } o; o.s = f2bf_raw(f); return o.b;
}
__device__ __forceinline__ unsigned fkey(float f) {           // order-preserving key
  union { float f; unsigned u; } c; c.f = f;
  return (c.u & 0x80000000u) ? ~c.u : (c.u | 0x80000000u);
}
__device__ __forceinline__ float fdec(unsigned k) {
  union { unsigned u; float f; } c;
  c.u = (k & 0x80000000u) ? (k & 0x7FFFFFFFu) : ~k;
  return c.f;
}

// ---------------------------------------------------------------------------
// Generic row-major GEMM: out[r, c] = f(sum_k A[r,k]*W[c,k] + bias[c]) + res[r,c]
// A fp32 (converted to bf16 in-register), W pre-converted bf16, 256x256.
// Workgroup tile 128x128; 8 waves; wave tile 32x64 (2x4 WMMA accumulators).
// ---------------------------------------------------------------------------
__global__ __launch_bounds__(256)
void gemm_rc(const float* __restrict__ A, const unsigned short* __restrict__ Wbf,
             const float* __restrict__ bias, const float* __restrict__ res,
             float* __restrict__ out, int do_gelu)
{
  const int lane = threadIdx.x & 31;
  const int wave = threadIdx.x >> 5;
  const int row0 = blockIdx.x * 128 + (wave >> 1) * 32;
  const int col0 = blockIdx.y * 128 + (wave & 1) * 64;
  const int half = lane >> 4;      // which 16-lane half
  const int l15  = lane & 15;

  v8f acc[2][4];
  #pragma unroll
  for (int i = 0; i < 2; i++)
    #pragma unroll
    for (int j = 0; j < 4; j++)
      #pragma unroll
      for (int r = 0; r < 8; r++) acc[i][j][r] = 0.0f;

  for (int kb = 0; kb < 256; kb += 32) {
    // A tiles: 16-bit A layout -> lane half h holds K {kh..kh+7} U {kh+16..kh+23}
    V16U a[2];
    #pragma unroll
    for (int mi = 0; mi < 2; mi++) {
      const float* ap = A + (size_t)(row0 + mi * 16 + l15) * 256 + kb + half * 8;
      float4 f0 = *(const float4*)(ap);
      float4 f1 = *(const float4*)(ap + 4);
      float4 f2 = *(const float4*)(ap + 16);
      float4 f3 = *(const float4*)(ap + 20);
      __builtin_prefetch(ap + 32, 0, 0);             // next K slice -> global_prefetch_b8
      a[mi].e[0]  = f2bf(f0.x); a[mi].e[1]  = f2bf(f0.y);
      a[mi].e[2]  = f2bf(f0.z); a[mi].e[3]  = f2bf(f0.w);
      a[mi].e[4]  = f2bf(f1.x); a[mi].e[5]  = f2bf(f1.y);
      a[mi].e[6]  = f2bf(f1.z); a[mi].e[7]  = f2bf(f1.w);
      a[mi].e[8]  = f2bf(f2.x); a[mi].e[9]  = f2bf(f2.y);
      a[mi].e[10] = f2bf(f2.z); a[mi].e[11] = f2bf(f2.w);
      a[mi].e[12] = f2bf(f3.x); a[mi].e[13] = f2bf(f3.y);
      a[mi].e[14] = f2bf(f3.z); a[mi].e[15] = f2bf(f3.w);
    }
    // B tiles: lane = column, half h holds K {8h..8h+7} U {8h+16..8h+23}
    V16U b[4];
    #pragma unroll
    for (int ni = 0; ni < 4; ni++) {
      const unsigned short* wp =
          Wbf + (size_t)(col0 + ni * 16 + l15) * 256 + kb + half * 8;
      const uint4* q = (const uint4*)wp;
      b[ni].q[0] = q[0];
      b[ni].q[1] = q[2];
    }
    #pragma unroll
    for (int mi = 0; mi < 2; mi++)
      #pragma unroll
      for (int ni = 0; ni < 4; ni++)
        acc[mi][ni] = __builtin_amdgcn_wmma_f32_16x16x32_bf16(
            false, a[mi].v, false, b[ni].v, (short)0, acc[mi][ni], false, false);
  }

  // Epilogue per documented C/D layout: VGPR r -> M = r + 8*half, N = l15
  #pragma unroll
  for (int mi = 0; mi < 2; mi++) {
    #pragma unroll
    for (int r = 0; r < 8; r++) {
      const int grow = row0 + mi * 16 + r + half * 8;
      const float* rrow = res ? res + (size_t)grow * 256 : nullptr;
      float* orow = out + (size_t)grow * 256;
      #pragma unroll
      for (int ni = 0; ni < 4; ni++) {
        const int gcol = col0 + ni * 16 + l15;
        float v = acc[mi][ni][r];
        if (bias) v += bias[gcol];
        if (do_gelu) v = 0.5f * v * (1.0f + erff(v * 0.70710678118654752f));
        if (res) v += rrow[gcol];
        orow[gcol] = v;
      }
    }
  }
}

// ---------------------------------------------------------------------------
// Small prep kernels
// ---------------------------------------------------------------------------
__global__ void cvt_bf16(const float* __restrict__ src,
                         unsigned short* __restrict__ dst, int n) {
  int i = blockIdx.x * 256 + threadIdx.x;
  if (i < n) dst[i] = f2bf_raw(src[i]);
}

// Amix[p, h*32+d] = sum_m mix_w[p, h*32+m] * qlat[h,m,d]  (fold scores into mix)
__global__ void build_amix(const float* __restrict__ mixw,
                           const float* __restrict__ qlat,
                           unsigned short* __restrict__ amix_bf) {
  int p = blockIdx.x, c = threadIdx.x;
  int h = c >> 5, d = c & 31;
  float acc = 0.f;
  #pragma unroll
  for (int m = 0; m < 32; m++)
    acc += mixw[p * 256 + h * 32 + m] * qlat[h * 1024 + m * 32 + d];
  amix_bf[p * 256 + c] = f2bf_raw(acc);
}

__global__ void init_stats(unsigned* encmax, float* encsum, float* z) {
  int i = blockIdx.x * 256 + threadIdx.x;   // 0..32767
  if (i < 1024) { encmax[i] = 0u; encsum[i] = 0.f; }
  z[i] = 0.f;
}

// ---------------------------------------------------------------------------
// Column softmax stats over N (enc softmax), mixed layout (b, n, p)
// ---------------------------------------------------------------------------
__global__ __launch_bounds__(256)
void colmax(const float* __restrict__ mixed, unsigned* __restrict__ encmax) {
  int b = blockIdx.y, chunk = blockIdx.x, p = threadIdx.x;
  const float* base = mixed + ((size_t)b * NSEQ + (size_t)chunk * 1024) * 256 + p;
  float m = -INFINITY;
  for (int n = 0; n < 1024; n++) m = fmaxf(m, base[(size_t)n * 256]);
  atomicMax(&encmax[b * 256 + p], fkey(m));
}

__global__ __launch_bounds__(256)
void colsumexp(const float* __restrict__ mixed, const unsigned* __restrict__ encmax,
               float* __restrict__ encsum) {
  int b = blockIdx.y, chunk = blockIdx.x, p = threadIdx.x;
  const float mx = fdec(encmax[b * 256 + p]);
  const float* base = mixed + ((size_t)b * NSEQ + (size_t)chunk * 1024) * 256 + p;
  float s = 0.f;
  for (int n = 0; n < 1024; n++) s += __expf(base[(size_t)n * 256] - mx);
  atomicAdd(&encsum[b * 256 + p], s);
}

// ---------------------------------------------------------------------------
// z_raw[b,h,m,d] = sum_n exp(mixed[b,n,hM+m]-max) * v[b,n,hD+d]   (LDS-staged)
// ---------------------------------------------------------------------------
__global__ __launch_bounds__(256)
void zaccum(const float* __restrict__ mixed, const float* __restrict__ v,
            const unsigned* __restrict__ encmax, float* __restrict__ z) {
  const int b = blockIdx.z, h = blockIdx.y, chunk = blockIdx.x;  // chunk: 512 n's
  __shared__ float es[8][32];
  __shared__ float vs[8][32];
  const int t = threadIdx.x;
  const int nl = t >> 5, c = t & 31;        // loader role
  const int m = t >> 3, d0 = (t & 7) * 4;   // accumulator role
  const float mx = fdec(encmax[b * 256 + h * 32 + c]);
  float a0 = 0.f, a1 = 0.f, a2 = 0.f, a3 = 0.f;
  const size_t rowbase = ((size_t)b * NSEQ + (size_t)chunk * 512) * 256 + h * 32;
  for (int g = 0; g < 64; g++) {
    const size_t off = rowbase + (size_t)(g * 8 + nl) * 256 + c;
    es[nl][c] = __expf(mixed[off] - mx);
    vs[nl][c] = v[off];
    __syncthreads();
    #pragma unroll
    for (int q = 0; q < 8; q++) {
      const float e = es[q][m];
      a0 += e * vs[q][d0 + 0]; a1 += e * vs[q][d0 + 1];
      a2 += e * vs[q][d0 + 2]; a3 += e * vs[q][d0 + 3];
    }
    __syncthreads();
  }
  float* zp = z + (((size_t)(b * 8 + h) * 32 + m) * 32 + d0);
  atomicAdd(zp + 0, a0); atomicAdd(zp + 1, a1);
  atomicAdd(zp + 2, a2); atomicAdd(zp + 3, a3);
}

// ---------------------------------------------------------------------------
// o[b,n,hD+d] = sum_m dec[b,h,m,n] * z_raw[b,h,m,d]/encsum[b,hM+m]
// One wave per head: shuffle-based softmax over M=32 (wave32).
// ---------------------------------------------------------------------------
__global__ __launch_bounds__(256)
void decode_o(const float* __restrict__ mixed, const float* __restrict__ z,
              const float* __restrict__ encsum, float* __restrict__ o) {
  const int b = blockIdx.y, chunk = blockIdx.x;          // 128 n's per block
  const int h = threadIdx.x >> 5, lane = threadIdx.x & 31;  // lane == d
  float zc[32];
  #pragma unroll
  for (int m = 0; m < 32; m++)
    zc[m] = z[((size_t)(b * 8 + h) * 32 + m) * 32 + lane] /
            encsum[b * 256 + h * 32 + m];
  const size_t base = ((size_t)b * NSEQ + (size_t)chunk * 128) * 256 + h * 32 + lane;
  for (int nl = 0; nl < 128; nl++) {
    float val = mixed[base + (size_t)nl * 256];   // lane acts as m for the softmax
    float mx = val;
    #pragma unroll
    for (int off = 16; off; off >>= 1) mx = fmaxf(mx, __shfl_xor(mx, off, 32));
    const float e = __expf(val - mx);
    float s = e;
    #pragma unroll
    for (int off = 16; off; off >>= 1) s += __shfl_xor(s, off, 32);
    float acc = 0.f;
    #pragma unroll
    for (int m = 0; m < 32; m++) acc += __shfl(e, m, 32) * zc[m];
    o[base + (size_t)nl * 256] = acc / s;
  }
}

// ---------------------------------------------------------------------------
extern "C" void kernel_launch(void* const* d_in, const int* in_sizes, int n_in,
                              void* d_out, int out_size, void* d_ws, size_t ws_size,
                              hipStream_t stream) {
  const float* x    = (const float*)d_in[0];
  const float* latq = (const float*)d_in[1];
  const float* k1w  = (const float*)d_in[2];  const float* k1b = (const float*)d_in[3];
  const float* ksw  = (const float*)d_in[4];  const float* ksb = (const float*)d_in[5];
  const float* k2w  = (const float*)d_in[6];  const float* k2b = (const float*)d_in[7];
  const float* v1w  = (const float*)d_in[8];  const float* v1b = (const float*)d_in[9];
  const float* vsw  = (const float*)d_in[10]; const float* vsb = (const float*)d_in[11];
  const float* v2w  = (const float*)d_in[12]; const float* v2b = (const float*)d_in[13];
  const float* mixw = (const float*)d_in[14];
  const float* outw = (const float*)d_in[15]; const float* outb = (const float*)d_in[16];
  float* out = (float*)d_out;

  char* ws = (char*)d_ws;
  const size_t SZF = (size_t)RTOT * 256 * sizeof(float);     // 128 MB
  float* hf0   = (float*)ws;              ws += SZF;
  float* hf1   = (float*)ws;              ws += SZF;
  float* mixed = (float*)ws;              ws += SZF;
  unsigned short* wbf    = (unsigned short*)ws; ws += (size_t)9 * 65536 * 2;
  unsigned short* amixbf = (unsigned short*)ws; ws += (size_t)65536 * 2;
  unsigned* encmax = (unsigned*)ws;       ws += 1024 * 4;
  float* encsum    = (float*)ws;          ws += 1024 * 4;
  float* zbuf      = (float*)ws;          ws += 32768 * 4;

  // --- prep: weights -> bf16, Amix fold, stats init ---
  cvt_bf16<<<256, 256, 0, stream>>>(k1w, wbf + 0 * 65536, 65536);
  cvt_bf16<<<512, 256, 0, stream>>>(ksw, wbf + 1 * 65536, 131072);   // ks0, ks1
  cvt_bf16<<<256, 256, 0, stream>>>(k2w, wbf + 3 * 65536, 65536);
  cvt_bf16<<<256, 256, 0, stream>>>(v1w, wbf + 4 * 65536, 65536);
  cvt_bf16<<<512, 256, 0, stream>>>(vsw, wbf + 5 * 65536, 131072);   // vs0, vs1
  cvt_bf16<<<256, 256, 0, stream>>>(v2w, wbf + 7 * 65536, 65536);
  cvt_bf16<<<256, 256, 0, stream>>>(outw, wbf + 8 * 65536, 65536);
  build_amix<<<256, 256, 0, stream>>>(mixw, latq, amixbf);
  init_stats<<<128, 256, 0, stream>>>(encmax, encsum, zbuf);

  const dim3 gg(1024, 2), gb(256);
  // --- k = ResidualMLP(x) ---
  gemm_rc<<<gg, gb, 0, stream>>>(x,   wbf + 0 * 65536, k1b,       x,   hf0, 1);
  gemm_rc<<<gg, gb, 0, stream>>>(hf0, wbf + 1 * 65536, ksb,       hf0, hf1, 1);
  gemm_rc<<<gg, gb, 0, stream>>>(hf1, wbf + 2 * 65536, ksb + 256, hf1, hf0, 1);
  gemm_rc<<<gg, gb, 0, stream>>>(hf0, wbf + 3 * 65536, k2b,       hf0, hf1, 0); // k
  // --- mixed = k @ Amix^T  (scores+mix fused) ---
  gemm_rc<<<gg, gb, 0, stream>>>(hf1, amixbf, nullptr, nullptr, mixed, 0);
  // --- v = ResidualMLP(x) ---
  gemm_rc<<<gg, gb, 0, stream>>>(x,   wbf + 4 * 65536, v1b,       x,   hf0, 1);
  gemm_rc<<<gg, gb, 0, stream>>>(hf0, wbf + 5 * 65536, vsb,       hf0, hf1, 1);
  gemm_rc<<<gg, gb, 0, stream>>>(hf1, wbf + 6 * 65536, vsb + 256, hf1, hf0, 1);
  gemm_rc<<<gg, gb, 0, stream>>>(hf0, wbf + 7 * 65536, v2b,       hf0, hf1, 0); // v
  // --- enc softmax stats over N ---
  colmax<<<dim3(32, 4), 256, 0, stream>>>(mixed, encmax);
  colsumexp<<<dim3(32, 4), 256, 0, stream>>>(mixed, encmax, encsum);
  // --- z = enc @ v (unnormalized) ---
  zaccum<<<dim3(64, 8, 4), 256, 0, stream>>>(mixed, hf1, encmax, zbuf);
  // --- o = dec . z  (into hf0) ---
  decode_o<<<dim3(256, 4), 256, 0, stream>>>(mixed, zbuf, encsum, hf0);
  // --- out = o @ out_w^T + out_b ---
  gemm_rc<<<gg, gb, 0, stream>>>(hf0, wbf + 8 * 65536, outb, nullptr, out, 0);
}
